// AMKPDModel_59450937311480
// MI455X (gfx1250) — compile-verified, matrix operand order checked by reading, not run on hardware
//
#include <hip/hip_runtime.h>
#include <hip/hip_bf16.h>
#include <cstdint>
#include <cstddef>

// ---------------------------------------------------------------------------
// Problem constants (match reference)
// ---------------------------------------------------------------------------
constexpr int B_  = 2;
constexpr int N_  = 1024;
constexpr int D_  = 512;
constexpr int H_  = 8;
constexpr int DH_ = 64;          // D_/H_
constexpr int K_  = 4;
constexpr int V_  = 32000;
constexpr int INNER_ = 1536;
constexpr int BN_ = B_ * N_;     // 2048 rows

typedef __attribute__((ext_vector_type(16))) __bf16 v16bf;
typedef __attribute__((ext_vector_type(8)))  float  v8f;

// ---------------------------------------------------------------------------
// CDNA5 async global->LDS copy (16B per lane), tracked by ASYNCcnt.
// Generic pointers to __shared__ have the LDS byte offset in their low 32 bits
// (ISA 10.2 aperture rules), which is what VDST of the async op expects.
// ---------------------------------------------------------------------------
__device__ __forceinline__ void async_copy16(void* lds, const void* gsrc) {
  uint32_t l = (uint32_t)(uintptr_t)lds;
  uint64_t g = (uint64_t)(uintptr_t)gsrc;
  asm volatile("global_load_async_to_lds_b128 %0, %1, off" ::"v"(l), "v"(g)
               : "memory");
}
__device__ __forceinline__ void wait_async() {
  asm volatile("s_wait_asynccnt 0" ::: "memory");
}

// ---------------------------------------------------------------------------
// WMMA fragment helpers (wave32, 16x16x32 bf16 layouts per CDNA5 ISA 7.12.2)
// A: lanes 0-15 -> row=lane, K={0..7,16..23}; lanes 16-31 -> row=lane-16,
//    K={8..15,24..31}. B: same K pattern per lane, column = lane&15.
// C/D: 8 f32 VGPRs; row = r + 8*(lane>=16), col = lane&15.
//
// B-side tiles are stored in LDS "fragment-swizzled": a 32(K)x16(N) block is
// 512 contiguous bf16 where lane L's 16 operand elements live at [L*16..L*16+15],
// so a fragment load is two ds_load_b128. frag_idx maps (ks,col) -> position.
// ---------------------------------------------------------------------------
__device__ __forceinline__ int frag_idx(int ks, int col) {
  int half = (ks >> 3) & 1;              // which lane half holds this k
  int i    = (ks & 7) | ((ks >> 1) & 8); // element index within lane
  return (half * 16 + col) * 16 + i;
}

__device__ __forceinline__ v16bf load_frag_A(const __bf16* t, int ld, int lane) {
  const int row = lane & 15;
  const int kb  = (lane >> 4) << 3;
  union { uint4 u[2]; v16bf v; } tmp;
  tmp.u[0] = *reinterpret_cast<const uint4*>(t + row * ld + kb);
  tmp.u[1] = *reinterpret_cast<const uint4*>(t + row * ld + kb + 16);
  return tmp.v;
}

__device__ __forceinline__ v16bf load_frag_sw(const __bf16* blk, int lane) {
  union { uint4 u[2]; v16bf v; } tmp;
  const uint4* p = reinterpret_cast<const uint4*>(blk + lane * 16);
  tmp.u[0] = p[0];
  tmp.u[1] = p[1];
  return tmp.v;
}

__device__ __forceinline__ v8f wmma_bf16(v16bf a, v16bf b, v8f c) {
  return __builtin_amdgcn_wmma_f32_16x16x32_bf16(false, a, false, b, (short)0, c,
                                                 false, false);
}

// ---------------------------------------------------------------------------
// Tiled bf16 GEMM: C[M,Nd] = epi(A[M,Kd] @ B)  (row-major)
//   BT=false: B row-major [Kd,Nd].  BT=true: B row-major [Nd,Kd] (B^T gemm).
// Epilogues:
//   0: outF = acc
//   2: outB = elu(acc)+1  (bf16)   [elu+1 == x>0? x+1 : exp(x)]
//   3: outF = acc; outB = bf16(acc)
//   4: outF = resid + softplus(dt)*acc
//   5: outF = resid + acc
// Tiles: BM=128, BN=64, BK=32; 256 threads = 8 waves in 2x4 (wr,wc); each wave
// owns four 16x16 subtiles: rows wr*64 + t*16 (t=0..3), cols wc*16..+15.
// ---------------------------------------------------------------------------
template <int EPI, bool BT>
__global__ __launch_bounds__(256) void gemm_bf16_k(
    const __bf16* __restrict__ A, const __bf16* __restrict__ Bm,
    float* __restrict__ outF, __bf16* __restrict__ outB,
    const float* __restrict__ resid, const float* __restrict__ dtp,
    int M, int Nd, int Kd) {
  constexpr int BM = 128, BNt = 64, BK = 32;
  __shared__ __align__(16) __bf16 sA[BM * BK];    // row-major A tile (8KB)
  __shared__ __align__(16) __bf16 sBs[4 * 512];   // 4 fragment-swizzled blocks

  const int tid  = threadIdx.x;
  const int lane = tid & 31;
  const int wave = tid >> 5;
  const int wr   = wave >> 2;
  const int wc   = wave & 3;
  const int tilesN = Nd / BNt;
  const int m0 = (blockIdx.x / tilesN) * BM;
  const int n0 = (blockIdx.x % tilesN) * BNt;

  v8f acc[4];
#pragma unroll
  for (int t = 0; t < 4; ++t) acc[t] = (v8f)0.0f;

  const int ar = tid >> 1;            // A stage: row 0..127
  const int ac = (tid & 1) * 16;      // A stage: col 0 or 16

  for (int k0 = 0; k0 < Kd; k0 += BK) {
    // ---- stage A tile via async global->LDS (2x16B per thread) ----
    {
      const __bf16* g = A + (size_t)(m0 + ar) * Kd + k0 + ac;
      async_copy16(&sA[ar * BK + ac], g);
      async_copy16(&sA[ar * BK + ac + 8], g + 8);
      if (k0 + BK < Kd) __builtin_prefetch(g + BK, 0, 0);
    }
    // ---- stage B tile, fragment-swizzled ----
    if (!BT) {
      int r = tid >> 3;               // ks 0..31
      int c = (tid & 7) * 8;          // col 0..56
      union { uint4 u; __bf16 h[8]; } tmp;
      tmp.u = *reinterpret_cast<const uint4*>(Bm + (size_t)(k0 + r) * Nd + n0 + c);
#pragma unroll
      for (int j = 0; j < 8; ++j) {
        int col = c + j;
        sBs[(col >> 4) * 512 + frag_idx(r, col & 15)] = tmp.h[j];
      }
    } else {
      int n  = tid >> 2;              // col 0..63
      int ck = (tid & 3) * 8;         // ks 0..24
      union { uint4 u; __bf16 h[8]; } tmp;
      tmp.u = *reinterpret_cast<const uint4*>(Bm + (size_t)(n0 + n) * Kd + k0 + ck);
#pragma unroll
      for (int j = 0; j < 8; ++j)
        sBs[(n >> 4) * 512 + frag_idx(ck + j, n & 15)] = tmp.h[j];
    }
    wait_async();
    __syncthreads();

    v16bf bf = load_frag_sw(&sBs[wc * 512], lane);
#pragma unroll
    for (int t = 0; t < 4; ++t) {
      v16bf af = load_frag_A(&sA[(wr * 64 + t * 16) * BK], BK, lane);
      acc[t] = wmma_bf16(af, bf, acc[t]);
    }
    __syncthreads();
  }

  const int colL = lane & 15;
  const int halfOff = (lane >> 4) * 8;
  float scale = 1.0f;
  if (EPI == 4) { float dv = dtp[0]; scale = log1pf(expf(dv)); }

#pragma unroll
  for (int t = 0; t < 4; ++t) {
#pragma unroll
    for (int r = 0; r < 8; ++r) {
      int row = m0 + wr * 64 + t * 16 + halfOff + r;
      int col = n0 + wc * 16 + colL;
      size_t idx = (size_t)row * Nd + col;
      float v = acc[t][r];
      if (EPI == 0) {
        outF[idx] = v;
      } else if (EPI == 2) {
        outB[idx] = (__bf16)(v > 0.0f ? v + 1.0f : expf(v));
      } else if (EPI == 3) {
        outF[idx] = v;
        outB[idx] = (__bf16)v;
      } else if (EPI == 4) {
        outF[idx] = resid[idx] + scale * v;
      } else if (EPI == 5) {
        outF[idx] = resid[idx] + v;
      }
    }
  }
}

// ---------------------------------------------------------------------------
// Fused linear attention: per (b,h,64-row tile):
//   S = PhiQ @ PhiK^T ; S = relu(S)^2 ; rowsum += S ; Attr += S(bf16) @ Vp
// Score matrix never touches HBM. PhiQ/PhiK/Vp stored [B*N, D] bf16
// (head h = columns h*64..h*64+63). Outputs Attr f32 [B*N,D], rowsum [B*N,H].
// K^T and V tiles live in LDS fragment-swizzled ([kstep][colblock] of 512).
// ---------------------------------------------------------------------------
__global__ __launch_bounds__(256) void attn_k(
    const __bf16* __restrict__ PhiQ, const __bf16* __restrict__ PhiK,
    const __bf16* __restrict__ Vp, float* __restrict__ Attr,
    float* __restrict__ rowsum) {
  const int tid  = threadIdx.x;
  const int lane = tid & 31;
  const int wave = tid >> 5;
  const int wr = wave >> 2, wc = wave & 3;
  const int tiles = N_ / 64;
  int bid = blockIdx.x;
  const int rt = bid % tiles; bid /= tiles;
  const int h  = bid % H_;
  const int b  = bid / H_;
  const int r0 = rt * 64;

  __shared__ __align__(16) __bf16 sQ[64 * 64];     // [row][d], A-side
  __shared__ __align__(16) __bf16 sKs[8 * 512];    // K^T swizzled (2 ksteps x 4 cb)
  __shared__ __align__(16) __bf16 sVs[8 * 512];    // V swizzled
  __shared__ __align__(16) __bf16 sS[64 * 64];     // staged bf16 scores, A-side
  __shared__ float sRS[64];

  {  // Q tile via async copies (2x16B per thread)
    int r = tid >> 2;
    int c = (tid & 3) * 16;
    const __bf16* g = PhiQ + (size_t)(b * N_ + r0 + r) * D_ + h * DH_ + c;
    async_copy16(&sQ[r * 64 + c], g);
    async_copy16(&sQ[r * 64 + c + 8], g + 8);
  }
  if (tid < 64) sRS[tid] = 0.0f;
  wait_async();

  v8f acc[2];
  acc[0] = (v8f)0.0f;
  acc[1] = (v8f)0.0f;
  float rs[16];
#pragma unroll
  for (int i = 0; i < 16; ++i) rs[i] = 0.0f;

  const int colL = lane & 15;
  const int halfOff = (lane >> 4) * 8;

  for (int m0 = 0; m0 < N_; m0 += 64) {
    __syncthreads();  // protect LDS reuse (and sQ/sRS visibility on iter 0)
    // ---- K^T tile: element (m,d): k=d, col=m -> swizzled scatter ----
#pragma unroll
    for (int ch = 0; ch < 2; ++ch) {
      int idx = tid * 2 + ch;  // 0..511 chunks of 8
      int m = idx >> 3;
      int c = (idx & 7) * 8;   // d base (stays inside one kstep)
      union { uint4 u; __bf16 hh[8]; } tmp;
      tmp.u = *reinterpret_cast<const uint4*>(
          PhiK + (size_t)(b * N_ + m0 + m) * D_ + h * DH_ + c);
      int kstep = c >> 5;
#pragma unroll
      for (int j = 0; j < 8; ++j) {
        int ks = (c & 31) + j;
        sKs[(kstep * 4 + (m >> 4)) * 512 + frag_idx(ks, m & 15)] = tmp.hh[j];
      }
    }
    // ---- V tile: element (m,d): k=m, col=d -> swizzled scatter ----
    {
      int m = tid >> 2;
      int c = (tid & 3) * 16;
      union { uint4 u[2]; __bf16 hh[16]; } tmp;
      const uint4* src = reinterpret_cast<const uint4*>(
          Vp + (size_t)(b * N_ + m0 + m) * D_ + h * DH_ + c);
      tmp.u[0] = src[0];
      tmp.u[1] = src[1];
      int kstep = m >> 5;
      int ks = m & 31;
#pragma unroll
      for (int j = 0; j < 16; ++j) {
        int d = c + j;
        sVs[(kstep * 4 + (d >> 4)) * 512 + frag_idx(ks, d & 15)] = tmp.hh[j];
      }
    }
    __syncthreads();

    // ---- S = Q @ K^T (dh=64 -> two WMMA K-steps) ----
    v8f s[2];
    s[0] = (v8f)0.0f;
    s[1] = (v8f)0.0f;
#pragma unroll
    for (int kk = 0; kk < 2; ++kk) {
      v16bf bf = load_frag_sw(&sKs[(kk * 4 + wc) * 512], lane);
#pragma unroll
      for (int t = 0; t < 2; ++t) {
        v16bf af = load_frag_A(&sQ[(wr * 32 + t * 16) * 64 + kk * 32], 64, lane);
        s[t] = wmma_bf16(af, bf, s[t]);
      }
    }
    // ---- relu^2, rowsum partials, stage bf16 scores ----
#pragma unroll
    for (int t = 0; t < 2; ++t) {
#pragma unroll
      for (int r = 0; r < 8; ++r) {
        float v = s[t][r];
        v = fmaxf(v, 0.0f);
        v = v * v;
        rs[t * 8 + r] += v;
        sS[(wr * 32 + t * 16 + halfOff + r) * 64 + wc * 16 + colL] = (__bf16)v;
      }
    }
    __syncthreads();

    // ---- Attr += S @ V ----
#pragma unroll
    for (int kk = 0; kk < 2; ++kk) {
      v16bf bf = load_frag_sw(&sVs[(kk * 4 + wc) * 512], lane);
#pragma unroll
      for (int t = 0; t < 2; ++t) {
        v16bf af = load_frag_A(&sS[(wr * 32 + t * 16) * 64 + kk * 32], 64, lane);
        acc[t] = wmma_bf16(af, bf, acc[t]);
      }
    }
  }

  __syncthreads();
#pragma unroll
  for (int t = 0; t < 2; ++t)
#pragma unroll
    for (int r = 0; r < 8; ++r)
      atomicAdd(&sRS[wr * 32 + t * 16 + halfOff + r], rs[t * 8 + r]);
  __syncthreads();
  if (tid < 64) rowsum[(size_t)(b * N_ + r0 + tid) * H_ + h] = sRS[tid];

#pragma unroll
  for (int t = 0; t < 2; ++t) {
#pragma unroll
    for (int r = 0; r < 8; ++r) {
      int row = r0 + wr * 32 + t * 16 + halfOff + r;
      Attr[(size_t)(b * N_ + row) * D_ + h * DH_ + wc * 16 + colL] = acc[t][r];
    }
  }
}

// ---------------------------------------------------------------------------
// LayerNorm helpers (D=512, one 256-thread block per row, 2 elems/thread)
// ---------------------------------------------------------------------------
__device__ __forceinline__ void ln_stats(float x0, float x1, float* red,
                                         int tid, float& mu, float& rinv) {
  red[tid] = x0 + x1;
  __syncthreads();
  for (int s = 128; s > 0; s >>= 1) {
    if (tid < s) red[tid] += red[tid + s];
    __syncthreads();
  }
  mu = red[0] / (float)D_;
  __syncthreads();
  float d0 = x0 - mu, d1 = x1 - mu;
  red[tid] = d0 * d0 + d1 * d1;
  __syncthreads();
  for (int s = 128; s > 0; s >>= 1) {
    if (tid < s) red[tid] += red[tid + s];
    __syncthreads();
  }
  float var = red[0] / (float)D_;
  __syncthreads();
  rinv = rsqrtf(var + 1e-5f);
}

__global__ __launch_bounds__(256) void embed_ln_k(
    const int* __restrict__ tok, const float* __restrict__ emb,
    const float* __restrict__ pos, const float* __restrict__ s,
    const float* __restrict__ bb, float* __restrict__ X) {
  const int row = blockIdx.x;  // b*N + n
  const int n = row % N_;
  const int tid = threadIdx.x;
  const int t = tok[row];
  __shared__ float red[256];
  float x0 = emb[(size_t)t * D_ + tid]       + pos[(size_t)n * D_ + tid];
  float x1 = emb[(size_t)t * D_ + tid + 256] + pos[(size_t)n * D_ + tid + 256];
  float mu, rinv;
  ln_stats(x0, x1, red, tid, mu, rinv);
  size_t base = (size_t)row * D_;
  X[base + tid]       = (x0 - mu) * rinv * s[tid]       + bb[tid];
  X[base + tid + 256] = (x1 - mu) * rinv * s[tid + 256] + bb[tid + 256];
}

__global__ __launch_bounds__(256) void ln_add_k(
    const float* __restrict__ Q, const float* __restrict__ s,
    const float* __restrict__ bb, const float* __restrict__ X,
    __bf16* __restrict__ Hc) {
  const int row = blockIdx.x;
  const int tid = threadIdx.x;
  __shared__ float red[256];
  size_t base = (size_t)row * D_;
  float x0 = Q[base + tid], x1 = Q[base + tid + 256];
  float mu, rinv;
  ln_stats(x0, x1, red, tid, mu, rinv);
  Hc[base + tid]       = (__bf16)((x0 - mu) * rinv * s[tid]       + bb[tid]       + X[base + tid]);
  Hc[base + tid + 256] = (__bf16)((x1 - mu) * rinv * s[tid + 256] + bb[tid + 256] + X[base + tid + 256]);
}

__global__ __launch_bounds__(256) void ln_bf_k(
    const float* __restrict__ Q, const float* __restrict__ s,
    const float* __restrict__ bb, __bf16* __restrict__ out) {
  const int row = blockIdx.x;
  const int tid = threadIdx.x;
  __shared__ float red[256];
  size_t base = (size_t)row * D_;
  float x0 = Q[base + tid], x1 = Q[base + tid + 256];
  float mu, rinv;
  ln_stats(x0, x1, red, tid, mu, rinv);
  out[base + tid]       = (__bf16)((x0 - mu) * rinv * s[tid]       + bb[tid]);
  out[base + tid + 256] = (__bf16)((x1 - mu) * rinv * s[tid + 256] + bb[tid + 256]);
}

__global__ __launch_bounds__(256) void ln_final_k(
    const float* __restrict__ Q, const float* __restrict__ s,
    const float* __restrict__ bb, float* __restrict__ outF,
    __bf16* __restrict__ outB) {
  const int row = blockIdx.x;
  const int tid = threadIdx.x;
  __shared__ float red[256];
  size_t base = (size_t)row * D_;
  float x0 = Q[base + tid], x1 = Q[base + tid + 256];
  float mu, rinv;
  ln_stats(x0, x1, red, tid, mu, rinv);
  float y0 = (x0 - mu) * rinv * s[tid]       + bb[tid];
  float y1 = (x1 - mu) * rinv * s[tid + 256] + bb[tid + 256];
  outF[base + tid] = y0;       outB[base + tid] = (__bf16)y0;
  outF[base + tid + 256] = y1; outB[base + tid + 256] = (__bf16)y1;
}

// ---------------------------------------------------------------------------
// Elementwise kernels
// ---------------------------------------------------------------------------
__global__ void cvt_bf16_k(const float* __restrict__ src, __bf16* __restrict__ dst,
                           size_t n) {
  size_t i = (size_t)blockIdx.x * 256 + threadIdx.x;
  if (i < n) dst[i] = (__bf16)src[i];
}

__global__ void select_tok_k(const int* __restrict__ a, const int* __restrict__ c,
                             const unsigned char* __restrict__ halted,
                             int* __restrict__ out) {
  int i = blockIdx.x * 256 + threadIdx.x;  // B*N
  int b = i / N_;
  out[i] = halted[b] ? a[i] : c[i];
}

__global__ void init_q_k(const float* __restrict__ ch, const float* __restrict__ ih,
                         const unsigned char* __restrict__ halted,
                         float* __restrict__ Q) {
  size_t i = (size_t)blockIdx.x * 256 + threadIdx.x;  // B*N*D
  int b = (int)(i / ((size_t)N_ * D_));
  int d = (int)(i % D_);
  Q[i] = halted[b] ? ih[d] : ch[i];
}

// m = Attr/(rowsum+1) - Vp   (bf16 output, [B*N, D] layout)
__global__ void make_m_k(const float* __restrict__ Attr,
                         const float* __restrict__ rowsum,
                         const float* __restrict__ Vpf,
                         __bf16* __restrict__ mb) {
  size_t i = (size_t)blockIdx.x * 256 + threadIdx.x;  // B*N*D
  size_t row = i / D_;
  int c = (int)(i % D_);
  int h = c >> 6;
  float w = rowsum[row * H_ + h] + 1.0f;
  mb[i] = (__bf16)(Attr[i] / w - Vpf[i]);
}

// Hf = silu(G)*U from GU [row, 2*INNER]
__global__ void silu_mul_k(const float* __restrict__ GU, float* __restrict__ Hf) {
  size_t i = (size_t)blockIdx.x * 256 + threadIdx.x;  // B*N*INNER
  size_t row = i / INNER_;
  int c = (int)(i % INNER_);
  float g = GU[row * (size_t)(2 * INNER_) + c];
  float u = GU[row * (size_t)(2 * INNER_) + INNER_ + c];
  Hf[i] = (g / (1.0f + expf(-g))) * u;
}

// depthwise conv-3 over N (zero pad) + bias + silu -> bf16
__global__ void dwconv_k(const float* __restrict__ Hf, const float* __restrict__ w,
                         const float* __restrict__ bias, __bf16* __restrict__ out) {
  size_t i = (size_t)blockIdx.x * 256 + threadIdx.x;  // B*N*INNER
  int c = (int)(i % INNER_);
  size_t bn = i / INNER_;
  int n = (int)(bn % N_);
  int b = (int)(bn / N_);
  float acc = bias[c];
#pragma unroll
  for (int j = 0; j < 3; ++j) {
    int nn = n + j - 1;
    if (nn >= 0 && nn < N_)
      acc += Hf[((size_t)(b * N_ + nn)) * INNER_ + c] * w[c * 3 + j];
  }
  out[i] = (__bf16)(acc / (1.0f + expf(-acc)));
}

// q_logits[b] = mean_n(Qn[b]) @ halt_w^T + halt_b
__global__ __launch_bounds__(256) void halt_k(
    const float* __restrict__ Qn, const float* __restrict__ hw,
    const float* __restrict__ hb, float* __restrict__ out) {
  const int b = blockIdx.x;
  const int tid = threadIdx.x;
  __shared__ float r0[256], r1[256];
  float p0 = 0.0f, p1 = 0.0f;
  for (int d = tid; d < D_; d += 256) {
    float s = 0.0f;
    for (int n = 0; n < N_; ++n) s += Qn[((size_t)(b * N_ + n)) * D_ + d];
    float m = s / (float)N_;
    p0 += m * hw[d];
    p1 += m * hw[D_ + d];
  }
  r0[tid] = p0; r1[tid] = p1;
  __syncthreads();
  for (int s = 128; s > 0; s >>= 1) {
    if (tid < s) { r0[tid] += r0[tid + s]; r1[tid] += r1[tid + s]; }
    __syncthreads();
  }
  if (tid == 0) {
    out[b * 2 + 0] = r0[0] + hb[0];
    out[b * 2 + 1] = r1[0] + hb[1];
  }
}

// ---------------------------------------------------------------------------
// Host launcher
// ---------------------------------------------------------------------------
extern "C" void kernel_launch(void* const* d_in, const int* in_sizes, int n_in,
                              void* d_out, int out_size, void* d_ws, size_t ws_size,
                              hipStream_t stream) {
  (void)in_sizes; (void)n_in; (void)out_size; (void)ws_size;

  const int*   inputs       = (const int*)d_in[0];
  const float* carry_hidden = (const float*)d_in[2];
  const unsigned char* carry_halted = (const unsigned char*)d_in[4];
  const int*   carry_inputs = (const int*)d_in[5];
  const float* init_hidden  = (const float*)d_in[7];
  const float* emb          = (const float*)d_in[8];
  const float* pos          = (const float*)d_in[9];
  const float* in_s  = (const float*)d_in[10];
  const float* in_b  = (const float*)d_in[11];
  const float* fin_s = (const float*)d_in[12];
  const float* fin_b = (const float*)d_in[13];
  const float* lm_w  = (const float*)d_in[14];
  const float* halt_w = (const float*)d_in[15];
  const float* halt_b = (const float*)d_in[16];
  const float* dt    = (const float*)d_in[17];
  const float* W_Q   = (const float*)d_in[18];
  const float* W_K   = (const float*)d_in[19];
  const float* W_V   = (const float*)d_in[20];
  const float* W_O   = (const float*)d_in[21];
  const float* W_up  = (const float*)d_in[22];
  const float* dw_w  = (const float*)d_in[23];
  const float* dw_b  = (const float*)d_in[24];
  const float* W_dn  = (const float*)d_in[25];
  const float* n1_s  = (const float*)d_in[26];
  const float* n1_b  = (const float*)d_in[27];
  const float* n2_s  = (const float*)d_in[28];
  const float* n2_b  = (const float*)d_in[29];

  // ---- workspace bump allocator (256B aligned) ----
  char* p = (char*)d_ws;
  auto alloc = [&](size_t bytes) -> void* {
    void* r = (void*)p;
    p += (bytes + 255) & ~(size_t)255;
    return r;
  };
  float*  X     = (float*) alloc((size_t)BN_ * D_ * 4);
  float*  Q     = (float*) alloc((size_t)BN_ * D_ * 4);
  __bf16* Hc    = (__bf16*)alloc((size_t)BN_ * D_ * 2);
  __bf16* PhiQ  = (__bf16*)alloc((size_t)BN_ * D_ * 2);
  __bf16* PhiK  = (__bf16*)alloc((size_t)BN_ * D_ * 2);
  __bf16* Vpb   = (__bf16*)alloc((size_t)BN_ * D_ * 2);
  float*  Vpf   = (float*) alloc((size_t)BN_ * D_ * 4);
  float*  Attr  = (float*) alloc((size_t)BN_ * D_ * 4);
  float*  rsum  = (float*) alloc((size_t)BN_ * H_ * 4);
  __bf16* mb    = (__bf16*)alloc((size_t)BN_ * D_ * 2);
  __bf16* Qn2   = (__bf16*)alloc((size_t)BN_ * D_ * 2);
  float*  GU    = (float*) alloc((size_t)BN_ * 2 * INNER_ * 4);
  float*  Hf    = (float*) alloc((size_t)BN_ * INNER_ * 4);
  __bf16* Hcv   = (__bf16*)alloc((size_t)BN_ * INNER_ * 2);
  float*  Qn    = (float*) alloc((size_t)BN_ * D_ * 4);
  __bf16* Qnb   = (__bf16*)alloc((size_t)BN_ * D_ * 2);
  int*    ntok  = (int*)   alloc((size_t)BN_ * 4);
  __bf16* Wqb   = (__bf16*)alloc((size_t)K_ * D_ * D_ * 2);
  __bf16* Wkb   = (__bf16*)alloc((size_t)K_ * D_ * D_ * 2);
  __bf16* Wvb   = (__bf16*)alloc((size_t)K_ * D_ * D_ * 2);
  __bf16* Wob   = (__bf16*)alloc((size_t)K_ * D_ * D_ * 2);
  __bf16* Wupb  = (__bf16*)alloc((size_t)K_ * D_ * 2 * INNER_ * 2);
  __bf16* Wdnb  = (__bf16*)alloc((size_t)K_ * INNER_ * D_ * 2);
  __bf16* lmwb  = (__bf16*)alloc((size_t)V_ * D_ * 2);

  auto cvt = [&](const float* s, __bf16* d, size_t n) {
    cvt_bf16_k<<<(unsigned)((n + 255) / 256), 256, 0, stream>>>(s, d, n);
  };
  cvt(W_Q,  Wqb,  (size_t)K_ * D_ * D_);
  cvt(W_K,  Wkb,  (size_t)K_ * D_ * D_);
  cvt(W_V,  Wvb,  (size_t)K_ * D_ * D_);
  cvt(W_O,  Wob,  (size_t)K_ * D_ * D_);
  cvt(W_up, Wupb, (size_t)K_ * D_ * 2 * INNER_);
  cvt(W_dn, Wdnb, (size_t)K_ * INNER_ * D_);
  cvt(lm_w, lmwb, (size_t)V_ * D_);

  // ---- carry reset + embedding ----
  select_tok_k<<<BN_ / 256, 256, 0, stream>>>(inputs, carry_inputs, carry_halted, ntok);
  init_q_k<<<(BN_ * D_) / 256, 256, 0, stream>>>(carry_hidden, init_hidden, carry_halted, Q);
  embed_ln_k<<<BN_, 256, 0, stream>>>(ntok, emb, pos, in_s, in_b, X);

  // ---- 8 block applications (H_CYCLES=2 x K=4) ----
  const int gD  = (BN_ / 128) * (D_ / 64);            // 128 tiles, D-out GEMMs
  const int gUp = (BN_ / 128) * ((2 * INNER_) / 64);  // 768 tiles
  for (int it = 0; it < 8; ++it) {
    const int k = it & 3;
    ln_add_k<<<BN_, 256, 0, stream>>>(Q, n1_s + k * D_, n1_b + k * D_, X, Hc);

    gemm_bf16_k<2, false><<<gD, 256, 0, stream>>>(
        Hc, Wqb + (size_t)k * D_ * D_, nullptr, PhiQ, nullptr, nullptr, BN_, D_, D_);
    gemm_bf16_k<2, false><<<gD, 256, 0, stream>>>(
        Hc, Wkb + (size_t)k * D_ * D_, nullptr, PhiK, nullptr, nullptr, BN_, D_, D_);
    gemm_bf16_k<3, false><<<gD, 256, 0, stream>>>(
        Hc, Wvb + (size_t)k * D_ * D_, Vpf, Vpb, nullptr, nullptr, BN_, D_, D_);

    attn_k<<<B_ * H_ * (N_ / 64), 256, 0, stream>>>(PhiQ, PhiK, Vpb, Attr, rsum);
    make_m_k<<<(BN_ * D_) / 256, 256, 0, stream>>>(Attr, rsum, Vpf, mb);

    // Qi = Q + softplus(dt[k]) * (m @ Wo)   (in-place into Q)
    gemm_bf16_k<4, false><<<gD, 256, 0, stream>>>(
        mb, Wob + (size_t)k * D_ * D_, Q, nullptr, Q, dt + k, BN_, D_, D_);

    ln_bf_k<<<BN_, 256, 0, stream>>>(Q, n2_s + k * D_, n2_b + k * D_, Qn2);
    gemm_bf16_k<0, false><<<gUp, 256, 0, stream>>>(
        Qn2, Wupb + (size_t)k * D_ * 2 * INNER_, GU, nullptr, nullptr, nullptr,
        BN_, 2 * INNER_, D_);
    silu_mul_k<<<(BN_ * INNER_) / 256, 256, 0, stream>>>(GU, Hf);
    dwconv_k<<<(BN_ * INNER_) / 256, 256, 0, stream>>>(
        Hf, dw_w + (size_t)k * INNER_ * 3, dw_b + (size_t)k * INNER_, Hcv);

    // Q = Qi + Hcv @ Wdn   (in-place into Q)
    gemm_bf16_k<5, false><<<gD, 256, 0, stream>>>(
        Hcv, Wdnb + (size_t)k * INNER_ * D_, Q, nullptr, Q, nullptr, BN_, D_, INNER_);
  }

  // ---- head ----
  ln_final_k<<<BN_, 256, 0, stream>>>(Q, fin_s, fin_b, Qn, Qnb);
  // logits = Qn @ lm_w^T  -> d_out[0 .. B*N*V)
  gemm_bf16_k<0, true><<<(BN_ / 128) * (V_ / 64), 256, 0, stream>>>(
      Qnb, lmwb, (float*)d_out, nullptr, nullptr, nullptr, BN_, V_, D_);
  // q_logits -> d_out[B*N*V ..]
  halt_k<<<B_, 256, 0, stream>>>(Qn, halt_w, halt_b,
                                 (float*)d_out + (size_t)BN_ * V_);
}
